// BudgetedMultiHeadAttention_41678362640618
// MI455X (gfx1250) — compile-verified, matrix operand order checked by reading, not run on hardware
//
#include <hip/hip_runtime.h>
#include <hip/hip_bf16.h>
#include <cstdint>
#include <cstddef>

// Problem constants (match reference)
#define BATCH   2
#define S_LEN   2048
#define D_MODEL 512
#define N_HEADS 8
#define DK      64      // D_MODEL / N_HEADS

typedef __attribute__((ext_vector_type(16))) _Float16 v16h;
typedef __attribute__((ext_vector_type(8)))  _Float16 v8h;
typedef __attribute__((ext_vector_type(8)))  float    v8f;
typedef __attribute__((ext_vector_type(4)))  float    v4f;

// ---------------------------------------------------------------------------
// WMMA wrapper: D(f32 16x16) = A(f16 16x32) x B(f16 32x16) + C
// ---------------------------------------------------------------------------
static __device__ __forceinline__ v8f wmma_f16(v16h a, v16h b, v8f c) {
    return __builtin_amdgcn_wmma_f32_16x16x32_f16(
        /*neg_a=*/false, a, /*neg_b=*/false, b,
        /*c_mod=*/(short)0, c, /*reuse_a=*/false, /*reuse_b=*/false);
}

// ---------------------------------------------------------------------------
// Fragment loader. A-matrix (and B^T, same layout) 16-bit 16x32:
//   lane<16 : row M = lane,    K = {0..7, 16..23}
//   lane>=16: row M = lane-16, K = {8..15, 24..31}
// rowptr must point at (row, kk) of a row-major matrix, 16B-alignable.
// ---------------------------------------------------------------------------
static __device__ __forceinline__ v16h frag_from_f16(const _Float16* rowptr, int lane) {
    const int base = (lane & 16) ? 8 : 0;
    v8h lo = *(const v8h*)(rowptr + base);
    v8h hi = *(const v8h*)(rowptr + base + 16);
    v16h r;
#pragma unroll
    for (int i = 0; i < 8; ++i) { r[i] = lo[i]; r[i + 8] = hi[i]; }
    return r;
}

// Row-wise butterfly reductions inside the 16-lane halves of a wave32
// (C/D layout: N = lane&15, so masks 1,2,4,8 reduce one matrix row).
static __device__ __forceinline__ float hmax16(float v) {
#pragma unroll
    for (int m = 1; m < 16; m <<= 1) v = fmaxf(v, __shfl_xor(v, m, 32));
    return v;
}
static __device__ __forceinline__ float hsum16(float v) {
#pragma unroll
    for (int m = 1; m < 16; m <<= 1) v += __shfl_xor(v, m, 32);
    return v;
}

// ---------------------------------------------------------------------------
// Kernel 0: one-shot f32 -> f16 conversion (x and the four weight matrices
// are reused 8..256x by the GEMMs; convert once, stream b128 in / b128 out).
// n8 = number of 8-element groups.
// ---------------------------------------------------------------------------
__global__ __launch_bounds__(256) void cvt_f16_kernel(
    const float* __restrict__ src, _Float16* __restrict__ dst, int n8)
{
    const int i = blockIdx.x * 256 + threadIdx.x;
    if (i >= n8) return;
    const v4f* p = (const v4f*)(src + (size_t)i * 8);
    v4f x0 = p[0], x1 = p[1];
    v8h o;
#pragma unroll
    for (int j = 0; j < 4; ++j) { o[j] = (_Float16)x0[j]; o[j + 4] = (_Float16)x1[j]; }
    *(v8h*)(dst + (size_t)i * 8) = o;
}

// ---------------------------------------------------------------------------
// Kernel 1: Q/K/V projections (all-f16 operands, f32 accumulate).
//   Q,K -> f16 (B,H,S,DK);  V -> f16 transposed (B,H,DK,S)
// 16x64 tile per wave; software-pipelined K-loop: next-iteration fragments
// are prefetched into separate registers so WMMAs see no WAR hazards.
// ---------------------------------------------------------------------------
__global__ __launch_bounds__(32) void qkv_kernel(
    const _Float16* __restrict__ x16,
    const _Float16* __restrict__ Wq16, const float* __restrict__ bq,
    const _Float16* __restrict__ Wk16, const float* __restrict__ bk,
    const _Float16* __restrict__ Wv16, const float* __restrict__ bv,
    _Float16* __restrict__ qh, _Float16* __restrict__ kh,
    _Float16* __restrict__ vTh)
{
    const int lane  = threadIdx.x;
    const int rl    = lane & 15;
    const int m0    = blockIdx.x * 16;
    const int n0    = blockIdx.y * 64;
    const int which = blockIdx.z;

    const _Float16* W;  const float* bi;
    if (which == 0)      { W = Wq16; bi = bq; }
    else if (which == 1) { W = Wk16; bi = bk; }
    else                 { W = Wv16; bi = bv; }

    const _Float16* arow = x16 + (size_t)(m0 + rl) * D_MODEL;
    const _Float16* brow = W   + (size_t)(n0 + rl) * D_MODEL;

    v16h a  = frag_from_f16(arow, lane);
    v16h b0 = frag_from_f16(brow, lane);
    v16h b1 = frag_from_f16(brow + 16 * D_MODEL, lane);
    v16h b2 = frag_from_f16(brow + 32 * D_MODEL, lane);
    v16h b3 = frag_from_f16(brow + 48 * D_MODEL, lane);

    v8f acc[4] = {};
    for (int kk = 0; kk < D_MODEL; kk += 32) {
        const int kn = (kk + 32) & (D_MODEL - 1);   // branch-free wrap
        v16h an  = frag_from_f16(arow + kn, lane);
        v16h bn0 = frag_from_f16(brow + kn, lane);
        v16h bn1 = frag_from_f16(brow + 16 * D_MODEL + kn, lane);
        v16h bn2 = frag_from_f16(brow + 32 * D_MODEL + kn, lane);
        v16h bn3 = frag_from_f16(brow + 48 * D_MODEL + kn, lane);
        acc[0] = wmma_f16(a, b0, acc[0]);
        acc[1] = wmma_f16(a, b1, acc[1]);
        acc[2] = wmma_f16(a, b2, acc[2]);
        acc[3] = wmma_f16(a, b3, acc[3]);
        a = an; b0 = bn0; b1 = bn1; b2 = bn2; b3 = bn3;
    }

    const int half8 = (lane & 16) ? 8 : 0;
#pragma unroll
    for (int sub = 0; sub < 4; ++sub) {
        const int n  = n0 + sub * 16 + rl;
        const float bn = bi[n];
        const int h  = n >> 6;       // n / DK
        const int d  = n & 63;       // n % DK
#pragma unroll
        for (int r = 0; r < 8; ++r) {
            const int m  = m0 + r + half8;           // global row in (B*S)
            const int b_ = m >> 11;                  // m / S_LEN
            const int s  = m & (S_LEN - 1);
            const float v = acc[sub][r] + bn;
            if (which == 2) {
                vTh[((size_t)(b_ * N_HEADS + h) * DK + d) * S_LEN + s] = (_Float16)v;
            } else {
                _Float16* dst = (which == 0) ? qh : kh;
                dst[((size_t)(b_ * N_HEADS + h) * S_LEN + s) * DK + d] = (_Float16)v;
            }
        }
    }
}

// ---------------------------------------------------------------------------
// Kernel 2: online softmax stats + entropy per row (flash pass 1).
// One wave per (b,h,16-query) tile; 32 keys/iter, two independent logits
// accumulator chains. Tracks m, l, t = sum e^(x-m)(x-m); H = log l - t/l.
// ---------------------------------------------------------------------------
__global__ __launch_bounds__(32) void stats_kernel(
    const _Float16* __restrict__ qh, const _Float16* __restrict__ kh,
    float* __restrict__ rowmax, float* __restrict__ rowl,
    float* __restrict__ hent)
{
    const int lane = threadIdx.x;
    const int rl   = lane & 15;
    const int q0   = blockIdx.x * 16;
    const int bh   = blockIdx.z * N_HEADS + blockIdx.y;

    const _Float16* qrow = qh + ((size_t)bh * S_LEN + q0 + rl) * DK;
    const v16h qa0 = frag_from_f16(qrow, lane);
    const v16h qa1 = frag_from_f16(qrow + 32, lane);

    float m[8], l[8], t[8];
#pragma unroll
    for (int r = 0; r < 8; ++r) { m[r] = -1e30f; l[r] = 0.f; t[r] = 0.f; }

    const _Float16* kbase = kh + (size_t)bh * S_LEN * DK;
    for (int c0 = 0; c0 < S_LEN; c0 += 32) {
        const _Float16* krowA = kbase + (size_t)(c0 + rl) * DK;
        const _Float16* krowB = kbase + (size_t)(c0 + 16 + rl) * DK;
        v16h ka0 = frag_from_f16(krowA, lane);
        v16h kb0 = frag_from_f16(krowB, lane);
        v16h ka1 = frag_from_f16(krowA + 32, lane);
        v16h kb1 = frag_from_f16(krowB + 32, lane);
        v8f accA = {}, accB = {};
        accA = wmma_f16(qa0, ka0, accA);
        accB = wmma_f16(qa0, kb0, accB);
        accA = wmma_f16(qa1, ka1, accA);
        accB = wmma_f16(qa1, kb1, accB);
#pragma unroll
        for (int sub = 0; sub < 2; ++sub) {
            const v8f& acc = sub ? accB : accA;
#pragma unroll
            for (int r = 0; r < 8; ++r) {
                const float x  = acc[r] * 0.125f;              // 1/sqrt(dk)
                const float cm = hmax16(x);
                const float nm = fmaxf(m[r], cm);
                const float pv = __expf(x - nm);
                const float sp = hsum16(pv);
                const float st = hsum16(pv * (x - nm));
                const float f  = __expf(m[r] - nm);
                t[r] = f * (t[r] + (m[r] - nm) * l[r]) + st;
                l[r] = f * l[r] + sp;
                m[r] = nm;
            }
        }
    }

    if (rl == 0) {
        const int half8 = (lane & 16) ? 8 : 0;
#pragma unroll
        for (int r = 0; r < 8; ++r) {
            const int s    = q0 + r + half8;
            const size_t i = (size_t)bh * S_LEN + s;
            rowmax[i] = m[r];
            rowl[i]   = l[r];
            hent[i]   = __logf(l[r]) - t[r] / l[r];
        }
    }
}

// ---------------------------------------------------------------------------
// Kernel 3: load -> budget.  One block per batch.
// ---------------------------------------------------------------------------
__global__ __launch_bounds__(256) void budget_kernel(
    const float* __restrict__ out_probs, const float* __restrict__ idf,
    const float* __restrict__ hent, float* __restrict__ bgt)
{
    __shared__ float sbuf[S_LEN];
    __shared__ float r1[256], r2[256];
    const int b = blockIdx.x, tid = threadIdx.x;

    // top-2 of out_probs[b, 0:1000]
    float t1 = -1e30f, t2 = -1e30f;
    for (int i = tid; i < 1000; i += 256) {
        float v = out_probs[b * 1000 + i];
        if (v > t1) { t2 = t1; t1 = v; } else if (v > t2) { t2 = v; }
    }
    r1[tid] = t1; r2[tid] = t2;
    __syncthreads();
    for (int s = 128; s > 0; s >>= 1) {
        if (tid < s) {
            float a1 = r1[tid], a2 = r2[tid], b1 = r1[tid + s], b2 = r2[tid + s];
            r1[tid] = fmaxf(a1, b1);
            r2[tid] = fmaxf(fminf(a1, b1), fmaxf(a2, b2));
        }
        __syncthreads();
    }
    const float margin = fmaxf(r1[0] - r2[0], 0.0f);
    // norm_minmax of a constant over S: (c-c)/max(0,1e-8) == 0
    const float lfm = 1.0f - ((margin - margin) / fmaxf(margin - margin, 1e-8f));
    __syncthreads();

    // head-mean entropy + minmax over S
    float mn = 1e30f, mx = -1e30f;
    for (int s = tid; s < S_LEN; s += 256) {
        float a = 0.f;
#pragma unroll
        for (int h = 0; h < N_HEADS; ++h)
            a += hent[(size_t)(b * N_HEADS + h) * S_LEN + s];
        const float v = a * (1.0f / N_HEADS);
        sbuf[s] = v;
        mn = fminf(mn, v); mx = fmaxf(mx, v);
    }
    r1[tid] = mn; r2[tid] = mx;
    __syncthreads();
    for (int s = 128; s > 0; s >>= 1) {
        if (tid < s) { r1[tid] = fminf(r1[tid], r1[tid + s]); r2[tid] = fmaxf(r2[tid], r2[tid + s]); }
        __syncthreads();
    }
    const float hmin = r1[0], hinv = 1.0f / fmaxf(r2[0] - r1[0], 1e-8f);
    __syncthreads();

    // L and its minmax
    mn = 1e30f; mx = -1e30f;
    for (int s = tid; s < S_LEN; s += 256) {
        const float Hn = (sbuf[s] - hmin) * hinv;
        const float L  = 0.4f * Hn + 0.4f * lfm + 0.2f * idf[b * S_LEN + s];
        sbuf[s] = L;
        mn = fminf(mn, L); mx = fmaxf(mx, L);
    }
    __syncthreads();
    r1[tid] = mn; r2[tid] = mx;
    __syncthreads();
    for (int s = 128; s > 0; s >>= 1) {
        if (tid < s) { r1[tid] = fminf(r1[tid], r1[tid + s]); r2[tid] = fmaxf(r2[tid], r2[tid + s]); }
        __syncthreads();
    }
    const float lmin = r1[0], linv = 1.0f / fmaxf(r2[0] - r1[0], 1e-8f);
    __syncthreads();

    for (int s = tid; s < S_LEN; s += 256)
        bgt[b * S_LEN + s] = 0.3f + 0.7f * (sbuf[s] - lmin) * linv;
}

// ---------------------------------------------------------------------------
// Kernel 4: flash pass 2. Recompute logits (interleaved WMMA chains),
// p = exp(x-m)*scale/l, stream scaled probs to d_out, f16 LDS round-trip
// to A-layout, attn@V with 4 independent WMMA accumulators.
// ---------------------------------------------------------------------------
__global__ __launch_bounds__(32) void attn_kernel(
    const _Float16* __restrict__ qh, const _Float16* __restrict__ kh,
    const _Float16* __restrict__ vTh,
    const float* __restrict__ rowmax, const float* __restrict__ rowl,
    const float* __restrict__ bgt,
    float* __restrict__ attn_out, _Float16* __restrict__ yh)
{
    __shared__ __align__(16) _Float16 ptile[16][32];

    const int lane  = threadIdx.x;
    const int rl    = lane & 15;
    const int half8 = (lane & 16) ? 8 : 0;
    const int q0    = blockIdx.x * 16;
    const int b     = blockIdx.z;
    const int bh    = b * N_HEADS + blockIdx.y;

    const _Float16* qrow = qh + ((size_t)bh * S_LEN + q0 + rl) * DK;
    const v16h qa0 = frag_from_f16(qrow, lane);
    const v16h qa1 = frag_from_f16(qrow + 32, lane);

    float mrow[8], c1[8];                 // c1 = scale / l  (softmax row_sum == 1)
#pragma unroll
    for (int r = 0; r < 8; ++r) {
        const int s = q0 + r + half8;
        mrow[r] = rowmax[(size_t)bh * S_LEN + s];
        const float sc = fminf(bgt[b * S_LEN + s], 1.0f);
        c1[r] = sc / rowl[(size_t)bh * S_LEN + s];
    }

    v8f yacc[4] = {};
    const _Float16* kbase = kh  + (size_t)bh * S_LEN * DK;
    const _Float16* vbase = vTh + (size_t)bh * DK * S_LEN;
    float* abase = attn_out + ((size_t)bh * S_LEN + q0) * S_LEN;

    for (int c0 = 0; c0 < S_LEN; c0 += 32) {
        const _Float16* krowA = kbase + (size_t)(c0 + rl) * DK;
        const _Float16* krowB = kbase + (size_t)(c0 + 16 + rl) * DK;
        v16h ka0 = frag_from_f16(krowA, lane);
        v16h kb0 = frag_from_f16(krowB, lane);
        v16h ka1 = frag_from_f16(krowA + 32, lane);
        v16h kb1 = frag_from_f16(krowB + 32, lane);
        v8f accA = {}, accB = {};
        accA = wmma_f16(qa0, ka0, accA);
        accB = wmma_f16(qa0, kb0, accB);
        accA = wmma_f16(qa1, ka1, accA);
        accB = wmma_f16(qa1, kb1, accB);
#pragma unroll
        for (int sub = 0; sub < 2; ++sub) {
            const v8f& acc = sub ? accB : accA;
            const int n0 = c0 + sub * 16;
#pragma unroll
            for (int r = 0; r < 8; ++r) {
                const float x = acc[r] * 0.125f;
                const float p = __expf(x - mrow[r]) * c1[r];
                const int row = r + half8;
                abase[(size_t)row * S_LEN + n0 + rl] = p;   // attn_probs output
                ptile[row][sub * 16 + rl] = (_Float16)p;
            }
        }
        __syncthreads();
        const v16h pa = frag_from_f16(&ptile[rl][0], lane); // C/D -> A layout via LDS
        __syncthreads();
#pragma unroll
        for (int dt = 0; dt < 4; ++dt) {
            const _Float16* vrow = vbase + (size_t)(dt * 16 + rl) * S_LEN + c0;
            v16h vb = frag_from_f16(vrow, lane);
            yacc[dt] = wmma_f16(pa, vb, yacc[dt]);
        }
    }

    _Float16* ybase = yh + ((size_t)bh * S_LEN + q0) * DK;
#pragma unroll
    for (int dt = 0; dt < 4; ++dt)
#pragma unroll
        for (int r = 0; r < 8; ++r)
            ybase[(size_t)(r + half8) * DK + dt * 16 + rl] = (_Float16)yacc[dt][r];
}

// ---------------------------------------------------------------------------
// Kernel 5: output projection  y = concat_heads(yh) @ Wo^T + bo  (fp32 out)
// 16x64 tile per wave, software-pipelined, 4 independent accumulator chains.
// ---------------------------------------------------------------------------
__global__ __launch_bounds__(32) void out_proj_kernel(
    const _Float16* __restrict__ yh, const _Float16* __restrict__ Wo16,
    const float* __restrict__ bo, float* __restrict__ y_out)
{
    const int lane = threadIdx.x;
    const int rl   = lane & 15;
    const int m0   = blockIdx.x * 16;
    const int n0   = blockIdx.y * 64;

    const int m  = m0 + rl;
    const int b_ = m >> 11;
    const int s  = m & (S_LEN - 1);
    const _Float16* brow = Wo16 + (size_t)(n0 + rl) * D_MODEL;

    // A rows live in head-blocked yh: k index c -> head c>>6, offset c&63
    const _Float16* abase = yh + ((size_t)(b_ * N_HEADS) * S_LEN + s) * DK;
    const size_t hstride = (size_t)S_LEN * DK;
#define AROW(kk2) (abase + ((kk2) >> 6) * hstride + ((kk2) & 63))

    v16h a  = frag_from_f16(AROW(0), lane);
    v16h b0 = frag_from_f16(brow, lane);
    v16h b1 = frag_from_f16(brow + 16 * D_MODEL, lane);
    v16h b2 = frag_from_f16(brow + 32 * D_MODEL, lane);
    v16h b3 = frag_from_f16(brow + 48 * D_MODEL, lane);

    v8f acc[4] = {};
    for (int kk = 0; kk < D_MODEL; kk += 32) {
        const int kn = (kk + 32) & (D_MODEL - 1);   // branch-free wrap
        v16h an  = frag_from_f16(AROW(kn), lane);
        v16h bn0 = frag_from_f16(brow + kn, lane);
        v16h bn1 = frag_from_f16(brow + 16 * D_MODEL + kn, lane);
        v16h bn2 = frag_from_f16(brow + 32 * D_MODEL + kn, lane);
        v16h bn3 = frag_from_f16(brow + 48 * D_MODEL + kn, lane);
        acc[0] = wmma_f16(a, b0, acc[0]);
        acc[1] = wmma_f16(a, b1, acc[1]);
        acc[2] = wmma_f16(a, b2, acc[2]);
        acc[3] = wmma_f16(a, b3, acc[3]);
        a = an; b0 = bn0; b1 = bn1; b2 = bn2; b3 = bn3;
    }
#undef AROW

    const int half8 = (lane & 16) ? 8 : 0;
#pragma unroll
    for (int sub = 0; sub < 4; ++sub) {
        const int n    = n0 + sub * 16 + rl;
        const float bn = bo[n];
#pragma unroll
        for (int r = 0; r < 8; ++r) {
            const int mm = m0 + r + half8;
            y_out[(size_t)mm * D_MODEL + n] = acc[sub][r] + bn;
        }
    }
}

// ---------------------------------------------------------------------------
// Host launcher
// ---------------------------------------------------------------------------
extern "C" void kernel_launch(void* const* d_in, const int* in_sizes, int n_in,
                              void* d_out, int out_size, void* d_ws, size_t ws_size,
                              hipStream_t stream) {
    (void)in_sizes; (void)n_in; (void)out_size; (void)ws_size;

    const float* x         = (const float*)d_in[0];
    const float* out_probs = (const float*)d_in[1];
    const float* idf       = (const float*)d_in[2];
    const float* Wq = (const float*)d_in[3];  const float* bq = (const float*)d_in[4];
    const float* Wk = (const float*)d_in[5];  const float* bk = (const float*)d_in[6];
    const float* Wv = (const float*)d_in[7];  const float* bv = (const float*)d_in[8];
    const float* Wo = (const float*)d_in[9];  const float* bo = (const float*)d_in[10];

    float* y_out    = (float*)d_out;                                  // (B,S,D)
    float* attn_out = y_out + (size_t)BATCH * S_LEN * D_MODEL;        // (B,H,S,S)

    const size_t nQKV = (size_t)BATCH * N_HEADS * S_LEN * DK;         // 2M elems
    const size_t nX   = (size_t)BATCH * S_LEN * D_MODEL;              // 2M elems
    const size_t nW   = (size_t)D_MODEL * D_MODEL;                    // 256K elems
    const size_t nRow = (size_t)BATCH * N_HEADS * S_LEN;              // 32K elems
    char* ws = (char*)d_ws;
    _Float16* qh   = (_Float16*)ws;                 ws += nQKV * sizeof(_Float16);
    _Float16* kh   = (_Float16*)ws;                 ws += nQKV * sizeof(_Float16);
    _Float16* vTh  = (_Float16*)ws;                 ws += nQKV * sizeof(_Float16);
    _Float16* yh   = (_Float16*)ws;                 ws += nQKV * sizeof(_Float16);
    _Float16* x16  = (_Float16*)ws;                 ws += nX   * sizeof(_Float16);
    _Float16* Wq16 = (_Float16*)ws;                 ws += nW   * sizeof(_Float16);
    _Float16* Wk16 = (_Float16*)ws;                 ws += nW   * sizeof(_Float16);
    _Float16* Wv16 = (_Float16*)ws;                 ws += nW   * sizeof(_Float16);
    _Float16* Wo16 = (_Float16*)ws;                 ws += nW   * sizeof(_Float16);
    float* rowmax  = (float*)ws;                    ws += nRow * sizeof(float);
    float* rowl    = (float*)ws;                    ws += nRow * sizeof(float);
    float* hent    = (float*)ws;                    ws += nRow * sizeof(float);
    float* bgt     = (float*)ws;                    ws += (size_t)BATCH * S_LEN * sizeof(float);

    const dim3 wave(32);
    const int nx8 = (int)(nX / 8), nw8 = (int)(nW / 8);
    cvt_f16_kernel<<<dim3(nx8 / 256), dim3(256), 0, stream>>>(x,  x16,  nx8);
    cvt_f16_kernel<<<dim3(nw8 / 256), dim3(256), 0, stream>>>(Wq, Wq16, nw8);
    cvt_f16_kernel<<<dim3(nw8 / 256), dim3(256), 0, stream>>>(Wk, Wk16, nw8);
    cvt_f16_kernel<<<dim3(nw8 / 256), dim3(256), 0, stream>>>(Wv, Wv16, nw8);
    cvt_f16_kernel<<<dim3(nw8 / 256), dim3(256), 0, stream>>>(Wo, Wo16, nw8);

    qkv_kernel<<<dim3((BATCH * S_LEN) / 16, D_MODEL / 64, 3), wave, 0, stream>>>(
        x16, Wq16, bq, Wk16, bk, Wv16, bv, qh, kh, vTh);
    stats_kernel<<<dim3(S_LEN / 16, N_HEADS, BATCH), wave, 0, stream>>>(
        qh, kh, rowmax, rowl, hent);
    budget_kernel<<<dim3(BATCH), dim3(256), 0, stream>>>(out_probs, idf, hent, bgt);
    attn_kernel<<<dim3(S_LEN / 16, N_HEADS, BATCH), wave, 0, stream>>>(
        qh, kh, vTh, rowmax, rowl, bgt, attn_out, yh);
    out_proj_kernel<<<dim3((BATCH * S_LEN) / 16, D_MODEL / 64), wave, 0, stream>>>(
        yh, Wo16, bo, y_out);
}
